// InteractionPPBlock_18751827214485
// MI455X (gfx1250) — compile-verified, matrix operand order checked by reading, not run on hardware
//
#include <hip/hip_runtime.h>
#include <math.h>

typedef __attribute__((ext_vector_type(16))) __bf16 v16bf;
typedef __attribute__((ext_vector_type(8)))  float  v8f;
typedef __attribute__((ext_vector_type(4)))  unsigned int u32x4;

#define H_DIM   128
#define INT_DIM 64
#define NR_DIM  6
#define SBF_DIM 42
#define BE_DIM  8
#define ASTRIDE 136   // 128 + 8 bf16 pad -> 272B row stride, rotates LDS banks

__device__ __forceinline__ unsigned short f32_to_bf16(float f) {
  unsigned int u = __float_as_uint(f);
  u += 0x7FFFu + ((u >> 16) & 1u);   // round-to-nearest-even
  return (unsigned short)(u >> 16);
}
__device__ __forceinline__ unsigned int pack_bf16x2(float lo, float hi) {
  return (unsigned int)f32_to_bf16(lo) | ((unsigned int)f32_to_bf16(hi) << 16);
}
__device__ __forceinline__ float silu_f(float x) { return x / (1.0f + __expf(-x)); }

union FragB { u32x4 q[2]; v16bf v; };

__device__ __forceinline__ v16bf load16bf(const unsigned short* p) {
  FragB f;
  const u32x4* q = (const u32x4*)p;
  f.q[0] = q[0];
  f.q[1] = q[1];
  return f.v;
}

// A fragment: 16x32 bf16 slice from LDS tile (row-major, stride ASTRIDE).
__device__ __forceinline__ v16bf frag_a(const unsigned short* As, int lane, int kbase) {
  int row  = lane & 15;
  int koff = kbase + ((lane >> 4) << 4);
  return load16bf(As + row * ASTRIDE + koff);
}

// B fragment: 32x16 bf16 from transposed weights WT[N][K] (K contiguous).
__device__ __forceinline__ v16bf frag_b(const unsigned short* WT, int Kdim, int lane,
                                        int nbase, int kbase) {
  int n    = nbase + (lane & 15);
  int koff = kbase + ((lane >> 4) << 4);
  return load16bf(WT + (size_t)n * Kdim + koff);
}

// Load full A tile (KDIM columns) once; reused across all 8 N-tiles.
template <int KDIM>
__device__ __forceinline__ void load_afrags(const unsigned short* As, int lane, v16bf* a) {
#pragma unroll
  for (int kt = 0; kt < KDIM / 32; ++kt) a[kt] = frag_a(As, lane, kt * 32);
}

// One 16x16 output tile: batched B loads, then dual-chain WMMA accumulation.
template <int KDIM>
__device__ __forceinline__ v8f gemm16(const v16bf* a, const unsigned short* WT, int lane,
                                      int nt, float binit) {
  constexpr int KT = KDIM / 32;
  v16bf b[KT];
#pragma unroll
  for (int kt = 0; kt < KT; ++kt) b[kt] = frag_b(WT, KDIM, lane, nt * 16, kt * 32);
  v8f acc0, acc1;
#pragma unroll
  for (int v = 0; v < 8; ++v) { acc0[v] = binit; acc1[v] = 0.0f; }
#pragma unroll
  for (int kt = 0; kt + 1 < KT; kt += 2) {
    acc0 = __builtin_amdgcn_wmma_f32_16x16x32_bf16(false, a[kt], false, b[kt], (short)0,
                                                   acc0, false, false);
    acc1 = __builtin_amdgcn_wmma_f32_16x16x32_bf16(false, a[kt + 1], false, b[kt + 1],
                                                   (short)0, acc1, false, false);
  }
  if (KT & 1)
    acc0 = __builtin_amdgcn_wmma_f32_16x16x32_bf16(false, a[KT - 1], false, b[KT - 1],
                                                   (short)0, acc0, false, false);
  if (KT >= 2) {
#pragma unroll
    for (int v = 0; v < 8; ++v) acc0[v] += acc1[v];
  }
  return acc0;
}

// C/D layout: VGPR v holds row m = v + 8*(lane>>4), col = nt*16 + (lane&15).
__device__ __forceinline__ void frag_to_lds(unsigned short* As, int lane, int nt, v8f c) {
  int col = nt * 16 + (lane & 15);
  int mb  = (lane >> 4) << 3;
#pragma unroll
  for (int v = 0; v < 8; ++v) As[(mb + v) * ASTRIDE + col] = f32_to_bf16(c[v]);
}

// ---------------- prep kernels ----------------

__global__ void wt_convert_kernel(const float* __restrict__ W, unsigned short* __restrict__ WT,
                                  int K, int N) {
  int o = blockIdx.x * blockDim.x + threadIdx.x;
  if (o < K * N) {
    int n = o / K, k = o - n * K;
    WT[o] = f32_to_bf16(W[(size_t)k * N + n]);
  }
}

__global__ void rbfc_kernel(const float* __restrict__ W1, const float* __restrict__ W2,
                            float* __restrict__ Wc) {
  int o = blockIdx.x * blockDim.x + threadIdx.x;
  if (o < NR_DIM * H_DIM) {
    int r = o / H_DIM, h = o - r * H_DIM;
    float s = 0.0f;
#pragma unroll
    for (int b = 0; b < BE_DIM; ++b) s += W1[r * BE_DIM + b] * W2[b * H_DIM + h];
    Wc[o] = s;
  }
}

__global__ void zero_kernel(float* __restrict__ p, long long n) {
  long long i = (long long)blockIdx.x * blockDim.x + threadIdx.x;
  if (i < n) p[i] = 0.0f;
}

// ---------------- stage 1: edge pre (ji, kj*rbf, down) ----------------

__global__ __launch_bounds__(128) void edge_pre_kernel(
    const float* __restrict__ x, const float* __restrict__ rbf,
    const float* __restrict__ Wrbfc, const unsigned short* __restrict__ WTji,
    const unsigned short* __restrict__ WTkj, const unsigned short* __restrict__ WTdown,
    const float* __restrict__ bji, const float* __restrict__ bkj,
    float* __restrict__ xji_out, float* __restrict__ xkjd, int E) {
  __shared__ alignas(16) unsigned short As[4][16 * ASTRIDE];
  __shared__ alignas(16) unsigned short Bs[4][16 * ASTRIDE];
  __shared__ float Rb[4][16 * NR_DIM];

  int lane = threadIdx.x & 31;
  int w    = threadIdx.x >> 5;
  int e0   = (blockIdx.x * 4 + w) * 16;
  if (e0 >= E) return;
  unsigned short* as = As[w];
  unsigned short* bs = Bs[w];
  float* rb = Rb[w];
  const bool full = (e0 + 16 <= E);

  // x tile [16 x 128] -> bf16 LDS (vectorized: float4 loads, 2x-packed b64 stores)
  if (full) {
#pragma unroll
    for (int i = 0; i < 16; ++i) {
      int f4 = i * 32 + lane;      // float4 index within tile (32 per row)
      int r = f4 >> 5, c4 = f4 & 31;
      float4 xv = *(const float4*)(x + (size_t)(e0 + r) * H_DIM + c4 * 4);
      uint2 p;
      p.x = pack_bf16x2(xv.x, xv.y);
      p.y = pack_bf16x2(xv.z, xv.w);
      *(uint2*)(as + r * ASTRIDE + c4 * 4) = p;
    }
  } else {
    for (int i = 0; i < 64; ++i) {
      int idx = i * 32 + lane;
      int r = idx >> 7, c = idx & 127;
      float xv = (e0 + r < E) ? x[(size_t)(e0 + r) * H_DIM + c] : 0.0f;
      as[r * ASTRIDE + c] = f32_to_bf16(xv);
    }
  }
  // rbf tile [16 x 6]
  for (int i = 0; i < 3; ++i) {
    int idx = i * 32 + lane;
    if (idx < 16 * NR_DIM) {
      int r = idx / NR_DIM, c = idx - r * NR_DIM;
      rb[idx] = (e0 + r < E) ? rbf[(size_t)(e0 + r) * NR_DIM + c] : 0.0f;
    }
  }

  int mb = (lane >> 4) << 3;
  v16bf a[4];
  load_afrags<H_DIM>(as, lane, a);

  // x_ji = silu(x @ W_ji + b_ji)  -> staged into d_out
  for (int nt = 0; nt < 8; ++nt) {
    int col = nt * 16 + (lane & 15);
    v8f c = gemm16<H_DIM>(a, WTji, lane, nt, bji[col]);
    if (full) {
#pragma unroll
      for (int v = 0; v < 8; ++v)
        xji_out[(size_t)(e0 + mb + v) * H_DIM + col] = silu_f(c[v]);
    } else {
      for (int v = 0; v < 8; ++v) {
        int e = e0 + mb + v;
        if (e < E) xji_out[(size_t)e * H_DIM + col] = silu_f(c[v]);
      }
    }
  }

  // x_kj = silu(x @ W_kj + b_kj) * rbf_p  -> bf16 into Bs tile
  for (int nt = 0; nt < 8; ++nt) {
    int col = nt * 16 + (lane & 15);
    v8f c = gemm16<H_DIM>(a, WTkj, lane, nt, bkj[col]);
    float wr[NR_DIM];
#pragma unroll
    for (int r = 0; r < NR_DIM; ++r) wr[r] = Wrbfc[r * H_DIM + col];
#pragma unroll
    for (int v = 0; v < 8; ++v) {
      int m = mb + v;
      float p = 0.0f;
#pragma unroll
      for (int r = 0; r < NR_DIM; ++r) p += wr[r] * rb[m * NR_DIM + r];
      c[v] = silu_f(c[v]) * p;
    }
    frag_to_lds(bs, lane, nt, c);
  }

  // xkjd = silu(x_kj @ W_down)
  v16bf a2[4];
  load_afrags<H_DIM>(bs, lane, a2);
  for (int nt = 0; nt < 4; ++nt) {
    int col = nt * 16 + (lane & 15);
    v8f c = gemm16<H_DIM>(a2, WTdown, lane, nt, 0.0f);
    if (full) {
#pragma unroll
      for (int v = 0; v < 8; ++v)
        xkjd[(size_t)(e0 + mb + v) * INT_DIM + col] = silu_f(c[v]);
    } else {
      for (int v = 0; v < 8; ++v) {
        int e = e0 + mb + v;
        if (e < E) xkjd[(size_t)e * INT_DIM + col] = silu_f(c[v]);
      }
    }
  }
}

// ---------------- stage 2: triplet gather/scatter ----------------

__global__ __launch_bounds__(256) void triplet_kernel(
    const float* __restrict__ sbf, const int* __restrict__ idx_kj,
    const int* __restrict__ idx_ji, const float* __restrict__ Wsbf1,
    const float* __restrict__ Wsbf2, const float* __restrict__ xkjd,
    float* __restrict__ agg, long long T) {
  __shared__ float W1s[SBF_DIM * BE_DIM];
  __shared__ float W2s[BE_DIM * INT_DIM];
  __shared__ float Us[8][32][BE_DIM];
  __shared__ int   KJ[8][32];
  __shared__ int   JI[8][32];

  int lane = threadIdx.x & 31;
  int w    = threadIdx.x >> 5;
  for (int i = threadIdx.x; i < SBF_DIM * BE_DIM; i += 256) W1s[i] = Wsbf1[i];
  for (int i = threadIdx.x; i < BE_DIM * INT_DIM; i += 256) W2s[i] = Wsbf2[i];
  __syncthreads();

  long long gw = (long long)blockIdx.x * 8 + w;
  long long nw = (long long)gridDim.x * 8;

  for (long long t0 = gw * 32; t0 < T; t0 += nw * 32) {
    long long t = t0 + lane;
    float u[BE_DIM];
#pragma unroll
    for (int j = 0; j < BE_DIM; ++j) u[j] = 0.0f;
    if (t < T) {
      const float* sr = sbf + t * SBF_DIM;
      for (int r = 0; r < SBF_DIM; ++r) {
        float s = sr[r];
#pragma unroll
        for (int j = 0; j < BE_DIM; ++j) u[j] += s * W1s[r * BE_DIM + j];
      }
      KJ[w][lane] = idx_kj[t];
      JI[w][lane] = idx_ji[t];
    } else {
      KJ[w][lane] = 0;
      JI[w][lane] = 0;
    }
#pragma unroll
    for (int j = 0; j < BE_DIM; ++j) Us[w][lane][j] = u[j];

    long long rem = T - t0;
    int nvalid = rem < 32 ? (int)rem : 32;
    for (int i = 0; i < nvalid; ++i) {
      int kj = KJ[w][i], ji = JI[w][i];
      float p0 = 0.0f, p1 = 0.0f;
#pragma unroll
      for (int j = 0; j < BE_DIM; ++j) {
        float uu = Us[w][i][j];  // LDS broadcast
        p0 += uu * W2s[j * INT_DIM + lane];
        p1 += uu * W2s[j * INT_DIM + lane + 32];
      }
      const float* g = xkjd + (size_t)kj * INT_DIM;   // L2-resident gather
      float m0 = p0 * g[lane];
      float m1 = p1 * g[lane + 32];
      float* ag = agg + (size_t)ji * INT_DIM;
      __hip_atomic_fetch_add(ag + lane, m0, __ATOMIC_RELAXED, __HIP_MEMORY_SCOPE_AGENT);
      __hip_atomic_fetch_add(ag + lane + 32, m1, __ATOMIC_RELAXED,
                             __HIP_MEMORY_SCOPE_AGENT);
    }
  }
}

// ---------------- stage 3: edge post (up, residual stack) ----------------

__global__ __launch_bounds__(128) void edge_post_kernel(
    const float* __restrict__ x, const float* __restrict__ agg,
    const unsigned short* __restrict__ WTup, const unsigned short* __restrict__ WTb0,
    const unsigned short* __restrict__ WTb1, const unsigned short* __restrict__ WTlin,
    const unsigned short* __restrict__ WTa00, const unsigned short* __restrict__ WTa01,
    const unsigned short* __restrict__ WTa10, const unsigned short* __restrict__ WTa11,
    const float* __restrict__ bb0, const float* __restrict__ bb1,
    const float* __restrict__ blin, const float* __restrict__ ba00,
    const float* __restrict__ ba01, const float* __restrict__ ba10,
    const float* __restrict__ ba11, float* __restrict__ out /* x_ji in, h out */, int E) {
  __shared__ alignas(16) unsigned short As[4][16 * ASTRIDE];
  __shared__ alignas(16) unsigned short Bs[4][16 * ASTRIDE];

  int lane = threadIdx.x & 31;
  int w    = threadIdx.x >> 5;
  int e0   = (blockIdx.x * 4 + w) * 16;
  if (e0 >= E) return;
  unsigned short* as = As[w];
  unsigned short* bs = Bs[w];
  int mb = (lane >> 4) << 3;
  const bool full = (e0 + 16 <= E);

  // agg tile [16 x 64] -> bf16 LDS (vectorized)
  if (full) {
#pragma unroll
    for (int i = 0; i < 8; ++i) {
      int f4 = i * 32 + lane;      // float4 index (16 per row)
      int r = f4 >> 4, c4 = f4 & 15;
      float4 av = *(const float4*)(agg + (size_t)(e0 + r) * INT_DIM + c4 * 4);
      uint2 p;
      p.x = pack_bf16x2(av.x, av.y);
      p.y = pack_bf16x2(av.z, av.w);
      *(uint2*)(as + r * ASTRIDE + c4 * 4) = p;
    }
  } else {
    for (int i = 0; i < 32; ++i) {
      int idx = i * 32 + lane;
      int r = idx >> 6, c = idx & 63;
      float v = (e0 + r < E) ? agg[(size_t)(e0 + r) * INT_DIM + c] : 0.0f;
      as[r * ASTRIDE + c] = f32_to_bf16(v);
    }
  }

  // h = x_ji + silu(agg @ W_up)
  v8f h[8];
  {
    v16bf a[2];
    load_afrags<INT_DIM>(as, lane, a);
    for (int nt = 0; nt < 8; ++nt) {
      int col = nt * 16 + (lane & 15);
      v8f c = gemm16<INT_DIM>(a, WTup, lane, nt, 0.0f);
#pragma unroll
      for (int v = 0; v < 8; ++v) {
        int e = e0 + mb + v;
        float xji = (full || e < E) ? out[(size_t)e * H_DIM + col] : 0.0f;
        h[nt][v] = xji + silu_f(c[v]);
      }
    }
  }

  // h += silu(silu(h @ W0 + b0) @ W1 + b1)   (ping-pong through As/Bs)
  auto resid = [&](const unsigned short* W0, const float* B0, const unsigned short* W1,
                   const float* B1) {
    for (int nt = 0; nt < 8; ++nt) frag_to_lds(as, lane, nt, h[nt]);
    v16bf a[4];
    load_afrags<H_DIM>(as, lane, a);
    for (int nt = 0; nt < 8; ++nt) {
      int col = nt * 16 + (lane & 15);
      v8f c = gemm16<H_DIM>(a, W0, lane, nt, B0[col]);
#pragma unroll
      for (int v = 0; v < 8; ++v) c[v] = silu_f(c[v]);
      frag_to_lds(bs, lane, nt, c);
    }
    load_afrags<H_DIM>(bs, lane, a);
    for (int nt = 0; nt < 8; ++nt) {
      int col = nt * 16 + (lane & 15);
      v8f c = gemm16<H_DIM>(a, W1, lane, nt, B1[col]);
#pragma unroll
      for (int v = 0; v < 8; ++v) h[nt][v] += silu_f(c[v]);
    }
  };

  resid(WTb0, bb0, WTb1, bb1);

  // h = silu(h @ W_lin + b_lin) + x
  {
    for (int nt = 0; nt < 8; ++nt) frag_to_lds(as, lane, nt, h[nt]);
    v16bf a[4];
    load_afrags<H_DIM>(as, lane, a);
    for (int nt = 0; nt < 8; ++nt) {
      int col = nt * 16 + (lane & 15);
      v8f c = gemm16<H_DIM>(a, WTlin, lane, nt, blin[col]);
#pragma unroll
      for (int v = 0; v < 8; ++v) {
        int e = e0 + mb + v;
        float xv = (full || e < E) ? x[(size_t)e * H_DIM + col] : 0.0f;
        h[nt][v] = silu_f(c[v]) + xv;
      }
    }
  }

  resid(WTa00, ba00, WTa01, ba01);
  resid(WTa10, ba10, WTa11, ba11);

  if (full) {
    for (int nt = 0; nt < 8; ++nt) {
      int col = nt * 16 + (lane & 15);
#pragma unroll
      for (int v = 0; v < 8; ++v)
        out[(size_t)(e0 + mb + v) * H_DIM + col] = h[nt][v];
    }
  } else {
    for (int nt = 0; nt < 8; ++nt) {
      int col = nt * 16 + (lane & 15);
      for (int v = 0; v < 8; ++v) {
        int e = e0 + mb + v;
        if (e < E) out[(size_t)e * H_DIM + col] = h[nt][v];
      }
    }
  }
}

// ---------------- host launcher ----------------

extern "C" void kernel_launch(void* const* d_in, const int* in_sizes, int n_in,
                              void* d_out, int out_size, void* d_ws, size_t ws_size,
                              hipStream_t stream) {
  const float* x      = (const float*)d_in[0];
  const float* rbf    = (const float*)d_in[1];
  const float* sbf    = (const float*)d_in[2];
  const int*   idx_kj = (const int*)d_in[3];
  const int*   idx_ji = (const int*)d_in[4];
  const float* W_rbf1 = (const float*)d_in[5];
  const float* W_rbf2 = (const float*)d_in[6];
  const float* W_sbf1 = (const float*)d_in[7];
  const float* W_sbf2 = (const float*)d_in[8];
  const float* W_kj   = (const float*)d_in[9];
  const float* b_kj   = (const float*)d_in[10];
  const float* W_ji   = (const float*)d_in[11];
  const float* b_ji   = (const float*)d_in[12];
  const float* W_down = (const float*)d_in[13];
  const float* W_up   = (const float*)d_in[14];
  const float* W_bef  = (const float*)d_in[15];
  const float* b_bef  = (const float*)d_in[16];
  const float* W_lin  = (const float*)d_in[17];
  const float* b_lin  = (const float*)d_in[18];
  const float* W_aft  = (const float*)d_in[19];
  const float* b_aft  = (const float*)d_in[20];
  float* out = (float*)d_out;

  const int E = in_sizes[0] / H_DIM;
  const long long T = in_sizes[3];

  unsigned char* ws = (unsigned char*)d_ws;
  size_t off = 0;
  auto take = [&](size_t bytes) -> void* {
    void* p = ws + off;
    off += (bytes + 255) & ~(size_t)255;
    return p;
  };
  unsigned short* WTji   = (unsigned short*)take(128 * 128 * 2);
  unsigned short* WTkj   = (unsigned short*)take(128 * 128 * 2);
  unsigned short* WTdown = (unsigned short*)take(128 * 64 * 2);
  unsigned short* WTup   = (unsigned short*)take(64 * 128 * 2);
  unsigned short* WTb0   = (unsigned short*)take(128 * 128 * 2);
  unsigned short* WTb1   = (unsigned short*)take(128 * 128 * 2);
  unsigned short* WTlin  = (unsigned short*)take(128 * 128 * 2);
  unsigned short* WTa00  = (unsigned short*)take(128 * 128 * 2);
  unsigned short* WTa01  = (unsigned short*)take(128 * 128 * 2);
  unsigned short* WTa10  = (unsigned short*)take(128 * 128 * 2);
  unsigned short* WTa11  = (unsigned short*)take(128 * 128 * 2);
  float* Wrbfc = (float*)take(NR_DIM * H_DIM * 4);
  float* xkjd  = (float*)take((size_t)E * INT_DIM * 4);
  float* agg   = (float*)take((size_t)E * INT_DIM * 4);

  const int n128 = 128 * 128, n64 = 128 * 64;
  const int g128 = (n128 + 255) / 256, g64 = (n64 + 255) / 256;
  wt_convert_kernel<<<g128, 256, 0, stream>>>(W_ji, WTji, 128, 128);
  wt_convert_kernel<<<g128, 256, 0, stream>>>(W_kj, WTkj, 128, 128);
  wt_convert_kernel<<<g64, 256, 0, stream>>>(W_down, WTdown, 128, 64);
  wt_convert_kernel<<<g64, 256, 0, stream>>>(W_up, WTup, 64, 128);
  wt_convert_kernel<<<g128, 256, 0, stream>>>(W_bef, WTb0, 128, 128);
  wt_convert_kernel<<<g128, 256, 0, stream>>>(W_bef + n128, WTb1, 128, 128);
  wt_convert_kernel<<<g128, 256, 0, stream>>>(W_lin, WTlin, 128, 128);
  wt_convert_kernel<<<g128, 256, 0, stream>>>(W_aft + 0 * n128, WTa00, 128, 128);
  wt_convert_kernel<<<g128, 256, 0, stream>>>(W_aft + 1 * n128, WTa01, 128, 128);
  wt_convert_kernel<<<g128, 256, 0, stream>>>(W_aft + 2 * n128, WTa10, 128, 128);
  wt_convert_kernel<<<g128, 256, 0, stream>>>(W_aft + 3 * n128, WTa11, 128, 128);
  rbfc_kernel<<<(NR_DIM * H_DIM + 255) / 256, 256, 0, stream>>>(W_rbf1, W_rbf2, Wrbfc);

  long long naggs = (long long)E * INT_DIM;
  zero_kernel<<<(int)((naggs + 255) / 256), 256, 0, stream>>>(agg, naggs);

  int tiles = (E + 15) / 16;
  int eblocks = (tiles + 3) / 4;
  edge_pre_kernel<<<eblocks, 128, 0, stream>>>(x, rbf, Wrbfc, WTji, WTkj, WTdown, b_ji,
                                               b_kj, out, xkjd, E);

  long long waves = (T + 31) / 32;
  int tblocks = (int)((waves + 7) / 8);
  triplet_kernel<<<tblocks, 256, 0, stream>>>(sbf, idx_kj, idx_ji, W_sbf1, W_sbf2, xkjd,
                                              agg, T);

  edge_post_kernel<<<eblocks, 128, 0, stream>>>(
      x, agg, WTup, WTb0, WTb1, WTlin, WTa00, WTa01, WTa10, WTa11, b_bef, b_bef + 128,
      b_lin, b_aft + 0, b_aft + 128, b_aft + 256, b_aft + 384, out, E);
}
